// EMACodebook_14723147890851
// MI455X (gfx1250) — compile-verified, matrix operand order checked by reading, not run on hardware
//
#include <hip/hip_runtime.h>

// VQ codebook quantization for MI455X (gfx1250, wave32, WMMA).
// dist(x,e) = ||x||^2 - 2 x.e + ||e||^2 ; argmin drops ||x||^2.
// GEMM (32768 x 8192 x 512) via v_wmma_f32_16x16x32_f16 with f32 accum.
// Tiles staged f32->f16 in LDS (269 KB of the 320 KB/WGP budget); whole
// problem is L2-resident (80 MB data vs 192 MB L2), so staging reads hit L2.
// Wave tile 32x64 = 2x4 WMMA tiles: 12 ds_load_b128 per 8 WMMAs.

#define DDIM   512
#define KCODES 8192
#define NROWS  32768
#define TM     128          // z rows per block (resident tile)
#define TN     128          // codebook rows per chunk
#define LDSTRIDE 520        // halves per LDS row (512 + 8 pad -> conflict-free b128)

typedef __attribute__((ext_vector_type(16))) _Float16 v16h;
typedef __attribute__((ext_vector_type(8)))  _Float16 v8h;
typedef __attribute__((ext_vector_type(4)))  _Float16 v4h;
typedef __attribute__((ext_vector_type(8)))  float    v8f;

// 16-bit A/B fragment (16x32 MxK / 32x16 KxN mirrored) per ISA 7.12.2:
// lanes 0-15: row/col = lane,      K = kk+{0..7}  and kk+{16..23}
// lanes 16-31: row/col = lane-16,  K = kk+{8..15} and kk+{24..31}
// -> two contiguous 16B chunks per lane = two ds_load_b128.
__device__ __forceinline__ v16h load_frag(const _Float16* ldsBase, int row0, int kk, int lane) {
  const int half = lane >> 4;
  const _Float16* p = ldsBase + (row0 + (lane & 15)) * LDSTRIDE + kk + half * 8;
  v8h lo = *(const v8h*)(p);
  v8h hi = *(const v8h*)(p + 16);
  return __builtin_shufflevector(lo, hi, 0,1,2,3,4,5,6,7,8,9,10,11,12,13,14,15);
}

__global__ void vq_zero_loss(float* __restrict__ loss) {
  if (threadIdx.x == 0 && blockIdx.x == 0) *loss = 0.0f;
}

// ||e_k||^2 for all 8192 codes: one wave per row, coalesced strided reads.
__global__ __launch_bounds__(256) void vq_enorm_kernel(const float* __restrict__ emb,
                                                       float* __restrict__ enorm) {
  const int lane = threadIdx.x & 31;
  const int row  = (blockIdx.x * blockDim.x + threadIdx.x) >> 5;
  if (row >= KCODES) return;
  const float* p = emb + (size_t)row * DDIM;
  float s = 0.0f;
  for (int j = lane; j < DDIM; j += 32) { float v = p[j]; s += v * v; }
  #pragma unroll
  for (int m = 16; m >= 1; m >>= 1) s += __shfl_xor(s, m, 32);
  if (lane == 0) enorm[row] = s;
}

__global__ __launch_bounds__(256) void vq_argmin_gemm_kernel(
    const float* __restrict__ z, const float* __restrict__ emb,
    const float* __restrict__ enorm, float* __restrict__ out_st,
    float* __restrict__ out_idx, float* __restrict__ out_loss) {

  __shared__ _Float16 zA[TM * LDSTRIDE];   // 130 KB: z tile (f16), block-resident
  __shared__ _Float16 eB[TN * LDSTRIDE];   // 130 KB: codebook chunk (f16)
  __shared__ float    minvS[TM][2];
  __shared__ int      miniS[TM][2];
  __shared__ int      idxS[TM];
  __shared__ float    lossS[8];

  const int tid   = threadIdx.x;
  const int lane  = tid & 31;
  const int wave  = tid >> 5;   // 8 waves
  const int wm    = wave & 3;   // 4 M-groups of 32 rows
  const int wn    = wave >> 2;  // 2 N-groups of 64 codes
  const int mBase = blockIdx.x * TM;

  // ---- stage z tile -> f16 LDS (coalesced float4 reads) ----
  {
    const float4* zg = (const float4*)(z + (size_t)mBase * DDIM);
    for (int i = tid; i < TM * (DDIM / 4); i += 256) {
      int r = i >> 7;                  // 128 float4 per row
      int c4 = i & 127;
      float4 v = zg[(size_t)r * (DDIM / 4) + c4];
      v4h h = { (_Float16)v.x, (_Float16)v.y, (_Float16)v.z, (_Float16)v.w };
      *(v4h*)(&zA[r * LDSTRIDE + c4 * 4]) = h;
    }
  }

  // running per-lane argmin state: entry (m,v) covers row
  //   mBase + wm*32 + m*16 + v + 8*(lane>>4); code varies with (lane&15, n, wn, chunk)
  float minv[2][8];
  int   mini[2][8];
  #pragma unroll
  for (int m = 0; m < 2; ++m)
    #pragma unroll
    for (int v = 0; v < 8; ++v) { minv[m][v] = 3.4e38f; mini[m][v] = 0; }

  const v8f vzero = {0.f,0.f,0.f,0.f,0.f,0.f,0.f,0.f};

  for (int cBase = 0; cBase < KCODES; cBase += TN) {
    __syncthreads();
    // ---- stage codebook chunk -> f16 LDS ----
    {
      const float4* eg = (const float4*)(emb + (size_t)cBase * DDIM);
      for (int i = tid; i < TN * (DDIM / 4); i += 256) {
        int r = i >> 7;
        int c4 = i & 127;
        float4 v = eg[(size_t)r * (DDIM / 4) + c4];
        v4h h = { (_Float16)v.x, (_Float16)v.y, (_Float16)v.z, (_Float16)v.w };
        *(v4h*)(&eB[r * LDSTRIDE + c4 * 4]) = h;
      }
    }
    if (cBase + TN < KCODES)   // warm L2/L0 for next chunk (global_prefetch_b8)
      __builtin_prefetch(emb + (size_t)(cBase + TN) * DDIM + tid * 64, 0, 0);
    __syncthreads();

    // ---- 32x64 wave tile: 2x4 WMMA tiles, K-loop over D ----
    v8f acc[2][4];
    #pragma unroll
    for (int m = 0; m < 2; ++m)
      #pragma unroll
      for (int n = 0; n < 4; ++n) acc[m][n] = vzero;

    #pragma unroll 2
    for (int kk = 0; kk < DDIM; kk += 32) {
      v16h a0 = load_frag(zA, wm * 32,      kk, lane);
      v16h a1 = load_frag(zA, wm * 32 + 16, kk, lane);
      v16h b0 = load_frag(eB, wn * 64,      kk, lane);
      v16h b1 = load_frag(eB, wn * 64 + 16, kk, lane);
      v16h b2 = load_frag(eB, wn * 64 + 32, kk, lane);
      v16h b3 = load_frag(eB, wn * 64 + 48, kk, lane);
      acc[0][0] = __builtin_amdgcn_wmma_f32_16x16x32_f16(false, a0, false, b0, (short)0, acc[0][0], false, false);
      acc[0][1] = __builtin_amdgcn_wmma_f32_16x16x32_f16(false, a0, false, b1, (short)0, acc[0][1], false, false);
      acc[0][2] = __builtin_amdgcn_wmma_f32_16x16x32_f16(false, a0, false, b2, (short)0, acc[0][2], false, false);
      acc[0][3] = __builtin_amdgcn_wmma_f32_16x16x32_f16(false, a0, false, b3, (short)0, acc[0][3], false, false);
      acc[1][0] = __builtin_amdgcn_wmma_f32_16x16x32_f16(false, a1, false, b0, (short)0, acc[1][0], false, false);
      acc[1][1] = __builtin_amdgcn_wmma_f32_16x16x32_f16(false, a1, false, b1, (short)0, acc[1][1], false, false);
      acc[1][2] = __builtin_amdgcn_wmma_f32_16x16x32_f16(false, a1, false, b2, (short)0, acc[1][2], false, false);
      acc[1][3] = __builtin_amdgcn_wmma_f32_16x16x32_f16(false, a1, false, b3, (short)0, acc[1][3], false, false);
    }

    // ---- fused epilogue: dist = ||e||^2 - 2*dot, update running argmin ----
    #pragma unroll
    for (int n = 0; n < 4; ++n) {
      const int code = cBase + wn * 64 + n * 16 + (lane & 15);
      const float en = enorm[code];
      #pragma unroll
      for (int m = 0; m < 2; ++m) {
        #pragma unroll
        for (int v = 0; v < 8; ++v) {
          float dist = en - 2.0f * acc[m][n][v];
          bool take = (dist < minv[m][v]) | ((dist == minv[m][v]) & (code < mini[m][v]));
          if (take) { minv[m][v] = dist; mini[m][v] = code; }
        }
      }
    }
  }

  // ---- argmin reduce: 16-lane butterflies within each half-wave ----
  #pragma unroll
  for (int m = 0; m < 2; ++m) {
    #pragma unroll
    for (int v = 0; v < 8; ++v) {
      float mv = minv[m][v];
      int   mi = mini[m][v];
      #pragma unroll
      for (int off = 8; off >= 1; off >>= 1) {
        float ov = __shfl_xor(mv, off, 32);
        int   oi = __shfl_xor(mi, off, 32);
        if ((ov < mv) | ((ov == mv) & (oi < mi))) { mv = ov; mi = oi; }
      }
      if ((lane & 15) == 0) {
        int rloc = wm * 32 + m * 16 + v + 8 * (lane >> 4);
        minvS[rloc][wn] = mv;
        miniS[rloc][wn] = mi;
      }
    }
  }
  __syncthreads();

  if (tid < TM) {
    float v0 = minvS[tid][0], v1 = minvS[tid][1];
    int   i0 = miniS[tid][0], i1 = miniS[tid][1];
    int sel = ((v1 < v0) | ((v1 == v0) & (i1 < i0))) ? i1 : i0;
    idxS[tid] = sel;
    out_idx[mBase + tid] = (float)sel;
  }
  __syncthreads();

  // ---- gather winning rows (f32 from L2), straight-through output + loss ----
  float lsum = 0.0f;
  for (int r = wave; r < TM; r += 8) {
    const int idx = idxS[r];
    const float4* er = (const float4*)(emb + (size_t)idx * DDIM);
    const float4* zr = (const float4*)(z + (size_t)(mBase + r) * DDIM);
    float4* orow = (float4*)(out_st + (size_t)(mBase + r) * DDIM);
    for (int c = lane; c < DDIM / 4; c += 32) {
      float4 e = er[c];
      float4 x = zr[c];
      orow[c] = e;    // stop_gradient(emb - z) + z == emb (forward value)
      float dx = x.x - e.x, dy = x.y - e.y, dz = x.z - e.z, dw = x.w - e.w;
      lsum += dx * dx + dy * dy + dz * dz + dw * dw;
    }
  }
  #pragma unroll
  for (int m = 16; m >= 1; m >>= 1) lsum += __shfl_xor(lsum, m, 32);
  if (lane == 0) lossS[wave] = lsum;
  __syncthreads();
  if (tid == 0) {
    float s = 0.0f;
    #pragma unroll
    for (int w = 0; w < 8; ++w) s += lossS[w];
    atomicAdd(out_loss, s * (0.25f / ((float)NROWS * (float)DDIM)));
  }
}

extern "C" void kernel_launch(void* const* d_in, const int* in_sizes, int n_in,
                              void* d_out, int out_size, void* d_ws, size_t ws_size,
                              hipStream_t stream) {
  (void)in_sizes; (void)n_in; (void)out_size; (void)ws_size;
  const float* z   = (const float*)d_in[0];   // [8,4096,512] f32
  const float* emb = (const float*)d_in[1];   // [8192,512]  f32

  float* out      = (float*)d_out;
  float* out_st   = out;                                   // [N, D]
  float* out_idx  = out + (size_t)NROWS * DDIM;            // [N]
  float* out_loss = out_idx + NROWS;                       // [1]

  float* enorm = (float*)d_ws;                             // 8192 floats scratch

  vq_zero_loss<<<1, 64, 0, stream>>>(out_loss);
  vq_enorm_kernel<<<(KCODES * 32) / 256, 256, 0, stream>>>(emb, enorm);
  vq_argmin_gemm_kernel<<<NROWS / TM, 256, 0, stream>>>(z, emb, enorm,
                                                        out_st, out_idx, out_loss);
}
